// GRUMPNN_43920335569128
// MI455X (gfx1250) — matrix operand order repaired and checked by reading, not compiled, not run on hardware
//
#include <hip/hip_runtime.h>

// ---------------------------------------------------------------------------
// CDNA5 (gfx1250) GRU-MPNN. bf16 WMMA (v_wmma_f32_16x16x32_bf16), fp32 accum.
// M-blocked: 32 rows (two 16x16 C tiles) per wave so each B fragment load
// feeds two WMMAs -> halves L2 weight traffic.
// ---------------------------------------------------------------------------

typedef __bf16 v16bf __attribute__((ext_vector_type(16)));
typedef float  v8f   __attribute__((ext_vector_type(8)));

#define HD 128   // node/edge hidden dim

static __device__ __forceinline__ unsigned short f2bf(float f) {
  unsigned u = __float_as_uint(f);
  u += 0x7FFFu + ((u >> 16) & 1u);        // round-to-nearest-even
  return (unsigned short)(u >> 16);
}

static __device__ __forceinline__ uint2 pack4bf(float4 v) {
  return make_uint2((unsigned)f2bf(v.x) | ((unsigned)f2bf(v.y) << 16),
                    (unsigned)f2bf(v.z) | ((unsigned)f2bf(v.w) << 16));
}

static __device__ __forceinline__ float sigmoidf(float x) {
  return 1.f / (1.f + __expf(-x));
}

static __device__ __forceinline__ v8f wmma_bf(v16bf a, v16bf b, v8f c) {
  // D = A(16x32 bf16) * B(32x16 bf16) + C(16x16 f32)
  return __builtin_amdgcn_wmma_f32_16x16x32_bf16(false, a, false, b,
                                                 (short)0, c, false, false);
}

union frag_u { v16bf v; uint4 q[2]; };

// A fragment (16x32 bf16). Lane L holds row m = L%16, half h = L/16.
// Elements 0..7  <- K = kt*32 + h*8 + [0..7]   (one ds b128)
// Elements 8..15 <- K = kt*32 + 16 + h*8 + [0..7]
static __device__ __forceinline__ v16bf load_a_frag(const unsigned short* row,
                                                    int kt, int lane) {
  const int h = (lane >> 4) & 1;
  frag_u u;
  u.q[0] = *(const uint4*)(row + kt * 32 + h * 8);
  u.q[1] = *(const uint4*)(row + kt * 32 + 16 + h * 8);
  return u.v;
}

// B fragment: weights pre-packed so each lane's 16 elements are contiguous.
static __device__ __forceinline__ v16bf load_b_frag(const unsigned short* blk,
                                                    int lane) {
  frag_u u;
  const uint4* p = (const uint4*)(blk + lane * 16);
  u.q[0] = p[0];
  u.q[1] = p[1];
  return u.v;
}

// ---------------------------------------------------------------------------
// Weight packing: W used as B[k][n]. wT=1 -> W stored [OUT][K] (GRU weights,
// y = x @ W.T). wT=0 -> W stored [K][OUT] (aW1). Block (kt,nt) holds a 32x16
// tile; lane L element i = B[kt*32 + (i/8)*16 + (L/16)*8 + i%8][nt*16 + L%16].
__global__ void pack_w_kernel(const float* __restrict__ W,
                              unsigned short* __restrict__ out,
                              int K, int OUT, int wT) {
  int idx = blockIdx.x * blockDim.x + threadIdx.x;
  if (idx >= K * OUT) return;
  int i    = idx & 15;
  int lane = (idx >> 4) & 31;
  int blk  = idx >> 9;
  int NT   = OUT >> 4;
  int kt   = blk / NT;
  int nt   = blk - kt * NT;
  int n = nt * 16 + (lane & 15);
  int k = kt * 32 + ((i >> 3) << 4) + ((lane >> 4) << 3) + (i & 7);
  float v = wT ? W[(long)n * K + k] : W[(long)k * OUT + n];
  out[idx] = f2bf(v);
}

// ---------------------------------------------------------------------------
__global__ void init_kernel(unsigned* __restrict__ nmax, float* __restrict__ nsum,
                            float* __restrict__ agg, int nN) {
  int i = blockIdx.x * blockDim.x + threadIdx.x;
  if (i < nN) { nmax[i] = 0u; nsum[i] = 0.f; }
  if (i < nN * HD) agg[i] = 0.f;
}

// order-preserving float->uint key for atomicMax-based segment max
static __device__ __forceinline__ unsigned fkey(float f) {
  unsigned u = __float_as_uint(f);
  return (u & 0x80000000u) ? ~u : (u | 0x80000000u);
}
static __device__ __forceinline__ float unfkey(unsigned k) {
  unsigned u = (k & 0x80000000u) ? (k & 0x7FFFFFFFu) : ~k;
  return __uint_as_float(u);
}

__global__ void logit_max_kernel(const float* __restrict__ logits,
                                 const int* __restrict__ dst,
                                 unsigned* __restrict__ nmax, int nE) {
  int e = blockIdx.x * blockDim.x + threadIdx.x;
  if (e < nE) atomicMax(&nmax[dst[e]], fkey(logits[e]));
}

__global__ void logit_exp_kernel(const float* __restrict__ logits,
                                 const int* __restrict__ dst,
                                 const unsigned* __restrict__ nmax,
                                 float* __restrict__ expE,
                                 float* __restrict__ nsum, int nE) {
  int e = blockIdx.x * blockDim.x + threadIdx.x;
  if (e < nE) {
    float x = __expf(logits[e] - unfkey(nmax[dst[e]]));
    expE[e] = x;
    atomicAdd(&nsum[dst[e]], x);
  }
}

__global__ void aggregate_kernel(const float* __restrict__ uef,
                                 const float* __restrict__ expE,
                                 const float* __restrict__ nsum,
                                 const int* __restrict__ dst,
                                 float* __restrict__ agg, int nE) {
  long i = (long)blockIdx.x * blockDim.x + threadIdx.x;
  int e = (int)(i >> 7);
  int d = (int)(i & 127);
  if (e < nE) {
    int dn = dst[e];
    float w = expE[e] / nsum[dn];
    atomicAdd(&agg[(long)dn * HD + d], uef[(long)e * HD + d] * w);
  }
}

// ---------------------------------------------------------------------------
// Edge GRU + attention-logit kernel. One wave32 per 32-edge tile (M-blocked:
// two 16-row A tiles share every B fragment).
__global__ __launch_bounds__(32) void edge_gru_kernel(
    const float* __restrict__ nf, const float* ef_in,
    const int* __restrict__ src, const int* __restrict__ dst,
    const unsigned short* __restrict__ Wih,   // [8 kt][24 nt] 32x16 blocks
    const unsigned short* __restrict__ Whh,   // [4 kt][24 nt]
    const unsigned short* __restrict__ Wa1,   // [4 kt][ 8 nt]
    const float* __restrict__ b_ih, const float* __restrict__ b_hh,
    const float* __restrict__ ab1, const float* __restrict__ aW2,
    const float* __restrict__ ab2,
    float* uef_out, float* __restrict__ logits, int nE) {
  const int lane = threadIdx.x;
  const int e0   = blockIdx.x * 32;

  __shared__ __align__(16) unsigned short xa[32][256];  // [src_h ; dst_h] bf16
  __shared__ __align__(16) unsigned short ha[32][128];  // ef bf16
  __shared__ __align__(16) unsigned short ue[32][128];  // uef bf16
  __shared__ float sred[32 * 16];
  __shared__ int   sidx[64];

  if (lane == 0) __builtin_prefetch(Wih, 0, 3);   // global_prefetch_b8

  {
    int ge = min(e0 + lane, nE - 1);
    sidx[lane]      = src[ge];
    sidx[32 + lane] = dst[ge];
  }
  __syncthreads();

  const float4* nf4 = (const float4*)nf;
  const float4* ef4 = (const float4*)ef_in;
  for (int e = 0; e < 32; ++e) {
    int ge = min(e0 + e, nE - 1);
    float4 vs = nf4[(long)sidx[e] * 32 + lane];
    float4 vd = nf4[(long)sidx[32 + e] * 32 + lane];
    float4 vh = ef4[(long)ge * 32 + lane];
    *(uint2*)&xa[e][lane * 4]       = pack4bf(vs);
    *(uint2*)&xa[e][128 + lane * 4] = pack4bf(vd);
    *(uint2*)&ha[e][lane * 4]       = pack4bf(vh);
  }
  __syncthreads();

  const int h  = lane >> 4;
  const int n  = lane & 15;
  const unsigned short* arow0 = &xa[n][0];        // A rows for tile 0
  const unsigned short* arow1 = &xa[16 + n][0];   // A rows for tile 1
  const unsigned short* hrow0 = &ha[n][0];
  const unsigned short* hrow1 = &ha[16 + n][0];

  for (int jt = 0; jt < 8; ++jt) {
    v8f ar0 = {}, az0 = {}, ain0 = {}, ahn0 = {};
    v8f ar1 = {}, az1 = {}, ain1 = {}, ahn1 = {};
#pragma unroll
    for (int kt = 0; kt < 8; ++kt) {        // K=256 over x_e
      v16bf a0 = load_a_frag(arow0, kt, lane);
      v16bf a1 = load_a_frag(arow1, kt, lane);
      v16bf br = load_b_frag(Wih + (kt * 24 + jt) * 512, lane);
      v16bf bz = load_b_frag(Wih + (kt * 24 + 8 + jt) * 512, lane);
      v16bf bn = load_b_frag(Wih + (kt * 24 + 16 + jt) * 512, lane);
      ar0  = wmma_bf(a0, br, ar0);  ar1  = wmma_bf(a1, br, ar1);
      az0  = wmma_bf(a0, bz, az0);  az1  = wmma_bf(a1, bz, az1);
      ain0 = wmma_bf(a0, bn, ain0); ain1 = wmma_bf(a1, bn, ain1);
    }
#pragma unroll
    for (int kt = 0; kt < 4; ++kt) {        // K=128 over ef
      v16bf a0 = load_a_frag(hrow0, kt, lane);
      v16bf a1 = load_a_frag(hrow1, kt, lane);
      v16bf br = load_b_frag(Whh + (kt * 24 + jt) * 512, lane);
      v16bf bz = load_b_frag(Whh + (kt * 24 + 8 + jt) * 512, lane);
      v16bf bn = load_b_frag(Whh + (kt * 24 + 16 + jt) * 512, lane);
      ar0  = wmma_bf(a0, br, ar0);  ar1  = wmma_bf(a1, br, ar1);
      az0  = wmma_bf(a0, bz, az0);  az1  = wmma_bf(a1, bz, az1);
      ahn0 = wmma_bf(a0, bn, ahn0); ahn1 = wmma_bf(a1, bn, ahn1);
    }
    const int c  = jt * 16 + n;
    float brz = b_ih[c] + b_hh[c];
    float bzz = b_ih[128 + c] + b_hh[128 + c];
    float bin = b_ih[256 + c];
    float bhn = b_hh[256 + c];
#pragma unroll
    for (int v = 0; v < 8; ++v) {           // C layout: m = v + 8h (+16), col c
#pragma unroll
      for (int t = 0; t < 2; ++t) {
        int m = v + 8 * h + 16 * t;
        long row = e0 + m;
        float rr = t ? ar1[v] : ar0[v];
        float zz = t ? az1[v] : az0[v];
        float ii = t ? ain1[v] : ain0[v];
        float hh = t ? ahn1[v] : ahn0[v];
        float gr = sigmoidf(rr + brz);
        float gz = sigmoidf(zz + bzz);
        float gn = tanhf(ii + bin + gr * (hh + bhn));
        if (row < nE) {
          float hprev = ef_in[row * HD + c];  // read-before-write: in-place ok
          float out   = (1.f - gz) * gn + gz * hprev;
          uef_out[row * HD + c] = out;
          ue[m][c] = f2bf(out);
        } else {
          ue[m][c] = 0;
        }
      }
    }
  }
  __syncthreads();

  // attention: logit = relu(uef @ aW1 + ab1) @ aW2 + ab2
  for (int i = lane; i < 32 * 16; i += 32) sred[i] = 0.f;
  __syncthreads();
  const unsigned short* urow0 = &ue[n][0];
  const unsigned short* urow1 = &ue[16 + n][0];
  for (int at = 0; at < 8; ++at) {
    v8f acc0 = {}, acc1 = {};
#pragma unroll
    for (int kt = 0; kt < 4; ++kt) {
      v16bf b = load_b_frag(Wa1 + (kt * 8 + at) * 512, lane);
      acc0 = wmma_bf(load_a_frag(urow0, kt, lane), b, acc0);
      acc1 = wmma_bf(load_a_frag(urow1, kt, lane), b, acc1);
    }
    float b1 = ab1[at * 16 + n];
    float w2 = aW2[at * 16 + n];
#pragma unroll
    for (int v = 0; v < 8; ++v) {
      float s0 = acc0[v] + b1;
      float s1 = acc1[v] + b1;
      s0 = s0 > 0.f ? s0 : 0.f;
      s1 = s1 > 0.f ? s1 : 0.f;
      sred[(v + 8 * h) * 16 + n]      += s0 * w2;   // distinct (m,n) per lane
      sred[(v + 8 * h + 16) * 16 + n] += s1 * w2;
    }
  }
  __syncthreads();
  {
    float sum = ab2[0];
#pragma unroll
    for (int i = 0; i < 16; ++i) sum += sred[lane * 16 + i];
    if (e0 + lane < nE) logits[e0 + lane] = sum;
  }
}

// ---------------------------------------------------------------------------
// Node GRU kernel: unf = GRU(agg, nf). One wave32 per 32-node tile.
__global__ __launch_bounds__(32) void node_gru_kernel(
    const float* __restrict__ agg, const float* nf_in,
    const unsigned short* __restrict__ Wih,   // [4 kt][24 nt]
    const unsigned short* __restrict__ Whh,   // [4 kt][24 nt]
    const float* __restrict__ b_ih, const float* __restrict__ b_hh,
    float* nf_out, int nN) {
  const int lane = threadIdx.x;
  const int n0   = blockIdx.x * 32;

  __shared__ __align__(16) unsigned short xa[32][128];
  __shared__ __align__(16) unsigned short ha[32][128];

  const float4* ag4 = (const float4*)agg;
  const float4* nf4 = (const float4*)nf_in;
  for (int e = 0; e < 32; ++e) {
    int gr = min(n0 + e, nN - 1);
    *(uint2*)&xa[e][lane * 4] = pack4bf(ag4[(long)gr * 32 + lane]);
    *(uint2*)&ha[e][lane * 4] = pack4bf(nf4[(long)gr * 32 + lane]);
  }
  __syncthreads();

  const int h = lane >> 4;
  const int n = lane & 15;
  const unsigned short* arow0 = &xa[n][0];
  const unsigned short* arow1 = &xa[16 + n][0];
  const unsigned short* hrow0 = &ha[n][0];
  const unsigned short* hrow1 = &ha[16 + n][0];

  for (int jt = 0; jt < 8; ++jt) {
    v8f ar0 = {}, az0 = {}, ain0 = {}, ahn0 = {};
    v8f ar1 = {}, az1 = {}, ain1 = {}, ahn1 = {};
#pragma unroll
    for (int kt = 0; kt < 4; ++kt) {
      v16bf a0 = load_a_frag(arow0, kt, lane);
      v16bf a1 = load_a_frag(arow1, kt, lane);
      v16bf br = load_b_frag(Wih + (kt * 24 + jt) * 512, lane);
      v16bf bz = load_b_frag(Wih + (kt * 24 + 8 + jt) * 512, lane);
      v16bf bn = load_b_frag(Wih + (kt * 24 + 16 + jt) * 512, lane);
      ar0  = wmma_bf(a0, br, ar0);  ar1  = wmma_bf(a1, br, ar1);
      az0  = wmma_bf(a0, bz, az0);  az1  = wmma_bf(a1, bz, az1);
      ain0 = wmma_bf(a0, bn, ain0); ain1 = wmma_bf(a1, bn, ain1);
      v16bf g0 = load_a_frag(hrow0, kt, lane);
      v16bf g1 = load_a_frag(hrow1, kt, lane);
      v16bf cr = load_b_frag(Whh + (kt * 24 + jt) * 512, lane);
      v16bf cz = load_b_frag(Whh + (kt * 24 + 8 + jt) * 512, lane);
      v16bf cn = load_b_frag(Whh + (kt * 24 + 16 + jt) * 512, lane);
      ar0  = wmma_bf(g0, cr, ar0);  ar1  = wmma_bf(g1, cr, ar1);
      az0  = wmma_bf(g0, cz, az0);  az1  = wmma_bf(g1, cz, az1);
      ahn0 = wmma_bf(g0, cn, ahn0); ahn1 = wmma_bf(g1, cn, ahn1);
    }
    const int c = jt * 16 + n;
    float brz = b_ih[c] + b_hh[c];
    float bzz = b_ih[128 + c] + b_hh[128 + c];
    float bin = b_ih[256 + c];
    float bhn = b_hh[256 + c];
#pragma unroll
    for (int v = 0; v < 8; ++v) {
#pragma unroll
      for (int t = 0; t < 2; ++t) {
        int m = v + 8 * h + 16 * t;
        long row = n0 + m;
        if (row < nN) {
          float rr = t ? ar1[v] : ar0[v];
          float zz = t ? az1[v] : az0[v];
          float ii = t ? ain1[v] : ain0[v];
          float hh = t ? ahn1[v] : ahn0[v];
          float gr = sigmoidf(rr + brz);
          float gz = sigmoidf(zz + bzz);
          float gn = tanhf(ii + bin + gr * (hh + bhn));
          float hprev = nf_in[row * HD + c];  // read-before-write: in-place ok
          nf_out[row * HD + c] = (1.f - gz) * gn + gz * hprev;
        }
      }
    }
  }
}

// ---------------------------------------------------------------------------
extern "C" void kernel_launch(void* const* d_in, const int* in_sizes, int n_in,
                              void* d_out, int out_size, void* d_ws, size_t ws_size,
                              hipStream_t stream) {
  const float* nf0   = (const float*)d_in[0];
  const float* ef0   = (const float*)d_in[1];
  const int*   src   = (const int*)d_in[2];
  const int*   dst   = (const int*)d_in[3];
  const float* eW_ih = (const float*)d_in[4];
  const float* eW_hh = (const float*)d_in[5];
  const float* eb_ih = (const float*)d_in[6];
  const float* eb_hh = (const float*)d_in[7];
  const float* nW_ih = (const float*)d_in[8];
  const float* nW_hh = (const float*)d_in[9];
  const float* nb_ih = (const float*)d_in[10];
  const float* nb_hh = (const float*)d_in[11];
  const float* aW1   = (const float*)d_in[12];
  const float* ab1   = (const float*)d_in[13];
  const float* aW2   = (const float*)d_in[14];
  const float* ab2   = (const float*)d_in[15];

  const int nE = in_sizes[2];
  const int nN = in_sizes[0] / HD;

  float* nf_buf = (float*)d_out;                     // [nN,128]
  float* ef_buf = (float*)d_out + (long)nN * HD;     // [nE,128]

  char* ws = (char*)d_ws;
  auto take = [&](size_t bytes) -> char* {
    char* p = ws;
    ws += (bytes + 255) & ~(size_t)255;
    return p;
  };
  unsigned short* Wih_e = (unsigned short*)take((size_t)8 * 24 * 512 * 2);
  unsigned short* Whh_e = (unsigned short*)take((size_t)4 * 24 * 512 * 2);
  unsigned short* Wih_n = (unsigned short*)take((size_t)4 * 24 * 512 * 2);
  unsigned short* Whh_n = (unsigned short*)take((size_t)4 * 24 * 512 * 2);
  unsigned short* Wa1_p = (unsigned short*)take((size_t)4 * 8 * 512 * 2);
  float*    logits = (float*)take((size_t)nE * 4);
  float*    expE   = (float*)take((size_t)nE * 4);
  unsigned* nmax   = (unsigned*)take((size_t)nN * 4);
  float*    nsum   = (float*)take((size_t)nN * 4);
  float*    agg    = (float*)take((size_t)nN * HD * 4);

  auto cdiv = [](long a, long b) { return (int)((a + b - 1) / b); };

  // one-time (per launch) weight packing into bf16 fragment layout
  pack_w_kernel<<<cdiv(256 * 384, 256), 256, 0, stream>>>(eW_ih, Wih_e, 256, 384, 1);
  pack_w_kernel<<<cdiv(128 * 384, 256), 256, 0, stream>>>(eW_hh, Whh_e, 128, 384, 1);
  pack_w_kernel<<<cdiv(128 * 384, 256), 256, 0, stream>>>(nW_ih, Wih_n, 128, 384, 1);
  pack_w_kernel<<<cdiv(128 * 384, 256), 256, 0, stream>>>(nW_hh, Whh_n, 128, 384, 1);
  pack_w_kernel<<<cdiv(128 * 128, 256), 256, 0, stream>>>(aW1, Wa1_p, 128, 128, 0);

  const int eTiles = cdiv(nE, 32);
  const int nTiles = cdiv(nN, 32);

  for (int it = 0; it < 2; ++it) {
    const float* nf_in = (it == 0) ? nf0 : nf_buf;
    const float* ef_in = (it == 0) ? ef0 : ef_buf;

    init_kernel<<<cdiv((long)nN * HD, 256), 256, 0, stream>>>(nmax, nsum, agg, nN);

    edge_gru_kernel<<<eTiles, 32, 0, stream>>>(
        nf_in, ef_in, src, dst, Wih_e, Whh_e, Wa1_p,
        eb_ih, eb_hh, ab1, aW2, ab2, ef_buf, logits, nE);

    logit_max_kernel<<<cdiv(nE, 256), 256, 0, stream>>>(logits, dst, nmax, nE);
    logit_exp_kernel<<<cdiv(nE, 256), 256, 0, stream>>>(logits, dst, nmax, expE, nsum, nE);
    aggregate_kernel<<<cdiv((long)nE * HD, 256), 256, 0, stream>>>(
        ef_buf, expE, nsum, dst, agg, nE);

    node_gru_kernel<<<nTiles, 32, 0, stream>>>(
        agg, nf_in, Wih_n, Whh_n, nb_ih, nb_hh, nf_buf, nN);
  }
}
